// EdgeDecoder_42941083025726
// MI455X (gfx1250) — matrix operand order repaired and checked by reading, not compiled, not run on hardware
//
#include <hip/hip_runtime.h>
#include <hip/hip_bf16.h>

// ---------------------------------------------------------------------------
// EdgeDecoder: out[e] = relu((z[u]*z[v]) @ W1 + b1) @ W2 + b2
//   z: [100000,128] f32, edge: [2,E] int32, W1:[128,128], b1:[128],
//   W2:[128,1], b2:[1], out: [E] f32, E = 640000.
//
// MI455X reasoning: z (51.2MB) fits in 192MB L2 -> gathers are L2-bound.
// GEMM work = 21.3 GFLOP; do it on matrix cores in f16 (f32 accumulate)
// via v_wmma_f32_16x16x32_f16 so the GEMM is far under the gather time.
// ---------------------------------------------------------------------------

typedef _Float16 h16 __attribute__((ext_vector_type(16)));
typedef _Float16 h8v __attribute__((ext_vector_type(8)));
typedef _Float16 h4v __attribute__((ext_vector_type(4)));
typedef float    f8v __attribute__((ext_vector_type(8)));

#define IN_CH  128
#define HID_CH 128
#define LDS_K  (IN_CH + 8)          // 136 halves = 272B row stride (16B aligned, bank-spread)
#define WAVES_PER_BLOCK   4
#define EDGES_PER_WAVE    16        // WMMA M
#define EDGES_PER_BLOCK   (WAVES_PER_BLOCK * EDGES_PER_WAVE)  // 64
#define BLOCK_THREADS     (WAVES_PER_BLOCK * 32)              // 128

__global__ __launch_bounds__(BLOCK_THREADS)
void edge_mlp_wmma_kernel(const float* __restrict__ z,
                          const int*   __restrict__ edge,   // [2, E] flat
                          const float* __restrict__ W1,     // [128,128] row-major (k,n)
                          const float* __restrict__ b1,     // [128]
                          const float* __restrict__ W2,     // [128,1]
                          const float* __restrict__ b2,     // [1]
                          float*       __restrict__ out,    // [E]
                          int nEdges)
{
    // W1 transposed to [n][k] in f16 so B fragments are contiguous K-runs.
    __shared__ _Float16 w1T[HID_CH][LDS_K];
    // Per-wave edge-feature tile x = z[u]*z[v], f16: [wave][edgeRow M][k]
    __shared__ _Float16 xA[WAVES_PER_BLOCK][EDGES_PER_WAVE][LDS_K];

    const int tid = threadIdx.x;

    // ---- Stage W1 (f32 -> f16, transpose) -------------------------------
    for (int i = tid; i < IN_CH * HID_CH; i += BLOCK_THREADS) {
        const int k = i >> 7;        // row of W1
        const int n = i & 127;       // col of W1
        w1T[n][k] = (_Float16)W1[i];
    }
    __syncthreads();

    const int wave = tid >> 5;
    const int lane = tid & 31;
    const int m    = lane & 15;      // WMMA row index within half-wave
    const int hi   = lane >> 4;      // 0: lanes 0-15, 1: lanes 16-31
    const int ebase = blockIdx.x * EDGES_PER_BLOCK + wave * EDGES_PER_WAVE;

    // ---- Load edge indices: lanes 0-15 -> src node, lanes 16-31 -> dst --
    int gidx = ebase + m;
    gidx = gidx < nEdges ? gidx : 0;                 // defensive clamp
    const int myIdx = (lane < 16) ? edge[gidx] : edge[nEdges + gidx];

    // ---- Gather + elementwise product, f32 -> f16 into LDS --------------
    // Each lane covers 4 channels (32 lanes x float4 = 128 ch) per edge.
    #pragma unroll 4
    for (int e = 0; e < EDGES_PER_WAVE; ++e) {
        const int u = __shfl(myIdx, e, 32);
        const int v = __shfl(myIdx, e + 16, 32);
        const float4 a4 = ((const float4*)(z + (size_t)u * IN_CH))[lane];
        const float4 c4 = ((const float4*)(z + (size_t)v * IN_CH))[lane];
        h4v p;
        p[0] = (_Float16)(a4.x * c4.x);
        p[1] = (_Float16)(a4.y * c4.y);
        p[2] = (_Float16)(a4.z * c4.z);
        p[3] = (_Float16)(a4.w * c4.w);
        *(h4v*)&xA[wave][e][lane * 4] = p;           // 8B store, aligned
    }
    __syncthreads();

    // ---- Build A fragments (ISA 7.12.2 16-bit A 16x32 layout) -----------
    // lane<16 : row M=lane,  K = kc*32 + {0..7, 16..23}
    // lane>=16: row M=m,     K = kc*32 + {8..15, 24..31}
    h16 afrag[4];
    #pragma unroll
    for (int kc = 0; kc < 4; ++kc) {
        const _Float16* arow = &xA[wave][m][0];
        const h8v lo  = *(const h8v*)(arow + kc * 32 + hi * 8);
        const h8v hi8 = *(const h8v*)(arow + kc * 32 + 16 + hi * 8);
        h16 a;
        #pragma unroll
        for (int t = 0; t < 8; ++t) { a[t] = lo[t]; a[t + 8] = hi8[t]; }
        afrag[kc] = a;
    }

    // ---- GEMM over 8 N-tiles, fused bias+ReLU+W2 epilogue ---------------
    // C/D layout: lane holds column N = nt*16 + m; VGPR r holds row hi*8+r.
    float partial[8];
    #pragma unroll
    for (int r = 0; r < 8; ++r) partial[r] = 0.0f;

    #pragma unroll
    for (int nt = 0; nt < 8; ++nt) {
        const int n = nt * 16 + m;
        f8v acc = {};
        #pragma unroll
        for (int kc = 0; kc < 4; ++kc) {
            // B fragment (32x16, f16): lane col N=n, K = kc*32 + hi*8 + {0..7, 16..23}
            const _Float16* brow = &w1T[n][0];
            const h8v blo = *(const h8v*)(brow + kc * 32 + hi * 8);
            const h8v bhi = *(const h8v*)(brow + kc * 32 + 16 + hi * 8);
            h16 b;
            #pragma unroll
            for (int t = 0; t < 8; ++t) { b[t] = blo[t]; b[t + 8] = bhi[t]; }
            acc = __builtin_amdgcn_wmma_f32_16x16x32_f16(
                      /*neg_a=*/false, afrag[kc],
                      /*neg_b=*/false, b,
                      /*c_mod=*/(short)0, acc,
                      /*reuse_a=*/false, /*reuse_b=*/false);
        }
        const float b1n = b1[n];
        const float w2n = W2[n];
        #pragma unroll
        for (int r = 0; r < 8; ++r) {
            float h = acc[r] + b1n;
            h = h > 0.0f ? h : 0.0f;
            partial[r] += h * w2n;     // fold ReLU + W2 dot into epilogue
        }
    }

    // ---- Reduce across the 16 lanes sharing each row set ----------------
    // (xor offsets < 16 keep lanes within their half-wave: rows match)
    #pragma unroll
    for (int off = 8; off >= 1; off >>= 1) {
        #pragma unroll
        for (int r = 0; r < 8; ++r)
            partial[r] += __shfl_xor(partial[r], off, 32);
    }

    if (m == 0) {
        const float b2v = b2[0];
        const int rowbase = hi * 8;   // lanes16-31 hold rows 8..15
        #pragma unroll
        for (int r = 0; r < 8; ++r) {
            const int e = ebase + rowbase + r;
            if (e < nEdges) out[e] = partial[r] + b2v;
        }
    }
}

extern "C" void kernel_launch(void* const* d_in, const int* in_sizes, int n_in,
                              void* d_out, int out_size, void* d_ws, size_t ws_size,
                              hipStream_t stream)
{
    const float* z    = (const float*)d_in[0];
    const int*   edge = (const int*)  d_in[1];   // [2, E]
    const float* W1   = (const float*)d_in[2];
    const float* b1   = (const float*)d_in[3];
    const float* W2   = (const float*)d_in[4];
    const float* b2   = (const float*)d_in[5];
    float* out = (float*)d_out;

    const int nEdges = in_sizes[1] / 2;                       // 640000
    const int blocks = (nEdges + EDGES_PER_BLOCK - 1) / EDGES_PER_BLOCK;  // 10000

    edge_mlp_wmma_kernel<<<blocks, BLOCK_THREADS, 0, stream>>>(
        z, edge, W1, b1, W2, b2, out, nEdges);
}
